// MultiHeadAttentionLayer_58755152609805
// MI455X (gfx1250) — matrix operand order repaired and checked by reading, not compiled, not run on hardware
//
#include <hip/hip_runtime.h>
#include <hip/hip_fp16.h>
#include <stdint.h>

typedef __attribute__((ext_vector_type(16))) _Float16 v16h;
typedef __attribute__((ext_vector_type(8)))  float    v8f;
typedef __attribute__((ext_vector_type(4)))  unsigned int v4u;
typedef __attribute__((ext_vector_type(8)))  int      v8i;
typedef __attribute__((ext_vector_type(4)))  int      v4i;

#if defined(__AMDGCN__) && __has_builtin(__builtin_amdgcn_tensor_load_to_lds)
#define USE_TDM 1
#else
#define USE_TDM 0
#endif

// ---------------------------------------------------------------------------
// WMMA fragment loaders; A/B use a contiguous k order (not the ISA table order, CDNA5 ISA)
// A 16x32 f16 (MxK): lane m=l&15, half=l>>4; VGPR v: k0=16h+2v
// B 32x16 f16 (KxN): lane n=l&15, half=l>>4; VGPR v: k0=16h+2v
// C/D 16x16 f32:     lane l, VGPR r -> (row r+8*(l>>4), col l&15)
// ---------------------------------------------------------------------------
__device__ __forceinline__ v16h frag_a(const _Float16* base, int ld) {
  const int lane = threadIdx.x & 31;
  const int m = lane & 15, hf = lane >> 4;
  union { v16h v; uint32_t u[8]; } f;
#pragma unroll
  for (int i = 0; i < 8; ++i) {
    const int k0 = 16 * hf + 2 * i;
    f.u[i] = *reinterpret_cast<const uint32_t*>(base + m * ld + k0);
  }
  return f.v;
}

// B source stored n-major: row n, k contiguous (pairs along k in one row)
__device__ __forceinline__ v16h frag_b(const _Float16* base, int ld) {
  const int lane = threadIdx.x & 31;
  const int n = lane & 15, hf = lane >> 4;
  union { v16h v; uint32_t u[8]; } f;
#pragma unroll
  for (int i = 0; i < 8; ++i) {
    const int k0 = 16 * hf + 2 * i;
    f.u[i] = *reinterpret_cast<const uint32_t*>(base + n * ld + k0);
  }
  return f.v;
}

// B source stored k-major (e.g. V tile [t][dh]): pairs along k are row pairs
__device__ __forceinline__ v16h frag_b_kmajor(const _Float16* base, int ld) {
  const int lane = threadIdx.x & 31;
  const int n = lane & 15, hf = lane >> 4;
  union { v16h v; _Float16 hx[16]; } f;
#pragma unroll
  for (int i = 0; i < 8; ++i) {
    const int k0 = 16 * hf + 2 * i;
    f.hx[2 * i]     = base[(k0)     * ld + n];
    f.hx[2 * i + 1] = base[(k0 + 1) * ld + n];
  }
  return f.v;
}

__device__ __forceinline__ v8f wmma_f32_16x16x32(v16h a, v16h b, v8f c) {
  return __builtin_amdgcn_wmma_f32_16x16x32_f16(false, a, false, b,
                                                (short)0, c, false, false);
}

#if USE_TDM
// LDS byte offset of a __shared__ object (generic -> AS3 -> integer)
typedef __attribute__((address_space(3))) const char* lds_cptr_t;
__device__ __forceinline__ unsigned lds_off(const void* p) {
  return (unsigned)(unsigned long long)(lds_cptr_t)p;
}

// Issue one TDM 2D tile load: tile 64 x 64 f16, tensor row stride 512 elems,
// LDS rows padded 128B -> 144B (pad_interval=4: every 32 DWORDs, pad_amount=3:
// 4 DWORDs), matching the LD = 72-element LDS row stride used by the frags.
__device__ __forceinline__ void tdm_load_tile_64x64_f16(unsigned lds_byte_off,
                                                        const _Float16* gsrc) {
  const unsigned long long ga = (unsigned long long)(uintptr_t)gsrc;
  v4u g0 = { 1u,                                   // count=1 (valid), no gather
             lds_byte_off,                         // lds_addr [63:32]
             (unsigned)ga,                         // global_addr [95:64]
             (unsigned)((ga >> 32) & 0x1FFFFFFu) | 0x80000000u };  // [120:96] | type=2
  v8i g1 = { (int)((1u << 16)     // data_size = 1 (2 bytes)
                 | (1u << 20)     // pad_enable
                 | (4u << 22)     // pad_interval: 32 DWORDs (128B)
                 | (3u << 25)),   // pad_amount: 4 DWORDs (16B)
             (int)(64u << 16),    // tensor_dim0 = 64 (low16 at [63:48])
             (int)(2048u << 16),  // tensor_dim0 hi=0, tensor_dim1 = 2048 (low16)
             (int)(64u << 16),    // tensor_dim1 hi=0, tile_dim0 = 64
             64,                  // tile_dim1 = 64, tile_dim2 = 0
             512, 0, 0 };         // tensor_dim0_stride = 512 elems
  v4i gz = { 0, 0, 0, 0 };
#if __clang_major__ >= 23
  v8i gz8 = { 0, 0, 0, 0, 0, 0, 0, 0 };
  __builtin_amdgcn_tensor_load_to_lds(g0, g1, gz, gz, gz8, 0);
#else
  __builtin_amdgcn_tensor_load_to_lds(g0, g1, gz, gz, 0);
#endif
}
#endif  // USE_TDM

// ---------------------------------------------------------------------------
// GEMM:  out[M,Nn] = X[M,K] @ W[Nn,K]^T + bias   (torch Linear semantics)
// Block tile 128x64, BK=32; 8 waves in a 4x2 grid; each wave owns 32x32.
// ---------------------------------------------------------------------------
template <typename AT, typename OT>
__global__ __launch_bounds__(256) void gemm_bias_wmma(
    const AT* __restrict__ X, const float* __restrict__ W,
    const float* __restrict__ bias, OT* __restrict__ out,
    int M, int K, int Nn) {
  constexpr int BM = 128, BN = 64, BK = 32, LDA = BK + 8;  // 80B row stride
  __shared__ _Float16 As[BM][LDA];
  __shared__ _Float16 Bs[BN][LDA];
  const int tid = threadIdx.x;
  const int w = tid >> 5;
  const int wm = w & 3, wn = w >> 2;  // 4(M) x 2(N) wave grid
  const int bM = blockIdx.y * BM, bN = blockIdx.x * BN;

  v8f acc[4] = {{}, {}, {}, {}};

  for (int kt = 0; kt < K; kt += BK) {
    __syncthreads();
    if constexpr (sizeof(AT) == 4) {
#pragma unroll
      for (int i = tid; i < BM * BK / 4; i += 256) {
        int r = i >> 3, c = (i & 7) << 2;
        float4 f4 = *reinterpret_cast<const float4*>(
            X + (size_t)(bM + r) * K + kt + c);
        As[r][c + 0] = (_Float16)f4.x; As[r][c + 1] = (_Float16)f4.y;
        As[r][c + 2] = (_Float16)f4.z; As[r][c + 3] = (_Float16)f4.w;
      }
    } else {
#pragma unroll
      for (int i = tid; i < BM * BK / 8; i += 256) {
        int r = i >> 2, c = (i & 3) << 3;
        *reinterpret_cast<uint4*>(&As[r][c]) =
            *reinterpret_cast<const uint4*>(X + (size_t)(bM + r) * K + kt + c);
      }
    }
#pragma unroll
    for (int i = tid; i < BN * BK / 4; i += 256) {
      int r = i >> 3, c = (i & 7) << 2;
      float4 f4 = *reinterpret_cast<const float4*>(
          W + (size_t)(bN + r) * K + kt + c);
      Bs[r][c + 0] = (_Float16)f4.x; Bs[r][c + 1] = (_Float16)f4.y;
      Bs[r][c + 2] = (_Float16)f4.z; Bs[r][c + 3] = (_Float16)f4.w;
    }
    __syncthreads();

    v16h a0 = frag_a(&As[wm * 32 + 0][0], LDA);
    v16h a1 = frag_a(&As[wm * 32 + 16][0], LDA);
    v16h b0 = frag_b(&Bs[wn * 32 + 0][0], LDA);
    v16h b1 = frag_b(&Bs[wn * 32 + 16][0], LDA);
    acc[0] = wmma_f32_16x16x32(a0, b0, acc[0]);
    acc[1] = wmma_f32_16x16x32(a0, b1, acc[1]);
    acc[2] = wmma_f32_16x16x32(a1, b0, acc[2]);
    acc[3] = wmma_f32_16x16x32(a1, b1, acc[3]);
  }

  const int lane = tid & 31;
  const int ro = 8 * (lane >> 4), co = lane & 15;
#pragma unroll
  for (int t = 0; t < 4; ++t) {
    const int ti = t >> 1, tj = t & 1;
    const int gc = bN + wn * 32 + tj * 16 + co;
    const float bv = bias[gc];
#pragma unroll
    for (int r = 0; r < 8; ++r) {
      const int gr = bM + wm * 32 + ti * 16 + r + ro;
      out[(size_t)gr * Nn + gc] = (OT)(acc[t][r] + bv);
    }
  }
}

// ---------------------------------------------------------------------------
// Fused flash attention: one block per (n, h, 128-row S tile); 8 waves, each
// owning a 16-row strip. K/V tiles DMA'd by the Tensor Data Mover (one
// tensor_load_to_lds per tile, TENSORcnt-tracked, hardware row padding to the
// 144B conflict-free LDS stride); scores + P@V via WMMA; online softmax in
// registers with shfl_xor reductions inside 16-lane halves.
// ---------------------------------------------------------------------------
__global__ __launch_bounds__(256) void flash_attn_wmma(
    const _Float16* __restrict__ q, const _Float16* __restrict__ k,
    const _Float16* __restrict__ v, _Float16* __restrict__ y) {
  constexpr int Sd = 2048, Dd = 512, DHd = 64, BR = 128, BC = 64;
  constexpr int LD = DHd + 8;   // 144B row stride (matches TDM pad)
  __shared__ _Float16 Qs[BR][LD];
  __shared__ _Float16 Ks[BC][LD];   // Ks[t][dh]
  __shared__ _Float16 Vs[BC][LD];   // Vs[t][dh] (k-major for P@V B-frags)
  __shared__ _Float16 Ps[8][16][LD];

  const int tid = threadIdx.x, lane = tid & 31, w = tid >> 5;
  const int bx = blockIdx.x;
  const int sblk = bx & 15, h = (bx >> 4) & 7, n = bx >> 7;

  const _Float16* qb = q + ((size_t)(n * Sd + sblk * BR)) * Dd + h * DHd;
  const _Float16* kb = k + (size_t)n * Sd * Dd + h * DHd;
  const _Float16* vb = v + (size_t)n * Sd * Dd + h * DHd;

  // ---- load Q tile once, pre-scaled by 1/sqrt(64) = 0.125 ----
  {
    const _Float16 sc = (_Float16)0.125f;
#pragma unroll
    for (int i = tid; i < BR * DHd / 8; i += 256) {
      int r = i >> 3, c = (i & 7) << 3;
      union { uint4 u4; _Float16 hx[8]; } u;
      u.u4 = *reinterpret_cast<const uint4*>(qb + (size_t)r * Dd + c);
#pragma unroll
      for (int j = 0; j < 8; ++j) u.hx[j] *= sc;
      *reinterpret_cast<uint4*>(&Qs[r][c]) = u.u4;
    }
  }
  __syncthreads();
  const v16h qf0 = frag_a(&Qs[w * 16][0], LD);       // K(dh) = 0..31
  const v16h qf1 = frag_a(&Qs[w * 16][0] + 32, LD);  // K(dh) = 32..63

  float m_run[8], l_run[8];
#pragma unroll
  for (int r = 0; r < 8; ++r) { m_run[r] = -1e30f; l_run[r] = 0.f; }
  v8f o[4] = {{}, {}, {}, {}};

  for (int tb = 0; tb < Sd / BC; ++tb) {
    __syncthreads();  // all waves done reading previous K/V tiles
#if USE_TDM
    if (tid < 32) {   // wave 0 drives the TDM; EXEC-independent wave-level DMA
      tdm_load_tile_64x64_f16(lds_off(&Ks[0][0]), kb + (size_t)tb * BC * Dd);
      tdm_load_tile_64x64_f16(lds_off(&Vs[0][0]), vb + (size_t)tb * BC * Dd);
      __builtin_amdgcn_s_wait_tensorcnt(0);
    }
#else
#pragma unroll
    for (int i = tid; i < BC * DHd / 8; i += 256) {
      int r = i >> 3, c = (i & 7) << 3;
      *reinterpret_cast<uint4*>(&Ks[r][c]) =
          *reinterpret_cast<const uint4*>(kb + (size_t)(tb * BC + r) * Dd + c);
      *reinterpret_cast<uint4*>(&Vs[r][c]) =
          *reinterpret_cast<const uint4*>(vb + (size_t)(tb * BC + r) * Dd + c);
    }
#endif
    if (tb + 1 < Sd / BC && tid < BC) {
      __builtin_prefetch(kb + (size_t)((tb + 1) * BC + tid) * Dd, 0, 1);
      __builtin_prefetch(vb + (size_t)((tb + 1) * BC + tid) * Dd, 0, 1);
    }
    __syncthreads();  // publish K/V tiles to all waves

    // ---- scores strip: 16 x 64 = 4 column tiles, K-dim = DH = 2 wmma steps ----
    v8f sacc[4] = {{}, {}, {}, {}};
#pragma unroll
    for (int c = 0; c < 4; ++c) {
      v16h kb0 = frag_b(&Ks[c * 16][0], LD);
      v16h kb1 = frag_b(&Ks[c * 16][0] + 32, LD);
      sacc[c] = wmma_f32_16x16x32(qf0, kb0, sacc[c]);
      sacc[c] = wmma_f32_16x16x32(qf1, kb1, sacc[c]);
    }

    // ---- online softmax (rows live in 16-lane halves) ----
#pragma unroll
    for (int r = 0; r < 8; ++r) {
      float mx = fmaxf(fmaxf(sacc[0][r], sacc[1][r]),
                       fmaxf(sacc[2][r], sacc[3][r]));
#pragma unroll
      for (int msk = 8; msk >= 1; msk >>= 1) mx = fmaxf(mx, __shfl_xor(mx, msk));
      const float mnew = fmaxf(m_run[r], mx);
      const float corr = __expf(m_run[r] - mnew);
      float rs = 0.f;
#pragma unroll
      for (int c = 0; c < 4; ++c) {
        float p = __expf(sacc[c][r] - mnew);
        sacc[c][r] = p;
        rs += p;
      }
#pragma unroll
      for (int msk = 8; msk >= 1; msk >>= 1) rs += __shfl_xor(rs, msk);
      l_run[r] = l_run[r] * corr + rs;
      m_run[r] = mnew;
#pragma unroll
      for (int j = 0; j < 4; ++j) o[j][r] *= corr;
    }

    // ---- reshape P: C-layout regs -> wave-private LDS strip -> A-frags ----
    const int rr0 = 8 * (lane >> 4), cc = lane & 15;
#pragma unroll
    for (int c = 0; c < 4; ++c)
#pragma unroll
      for (int r = 0; r < 8; ++r)
        Ps[w][r + rr0][c * 16 + cc] = (_Float16)sacc[c][r];

    v16h pf0 = frag_a(&Ps[w][0][0], LD);        // K(t) = 0..31
    v16h pf1 = frag_a(&Ps[w][0][0] + 32, LD);   // K(t) = 32..63
#pragma unroll
    for (int j = 0; j < 4; ++j) {
      v16h vb0 = frag_b_kmajor(&Vs[0][j * 16], LD);    // t = 0..31
      v16h vb1 = frag_b_kmajor(&Vs[32][j * 16], LD);   // t = 32..63
      o[j] = wmma_f32_16x16x32(pf0, vb0, o[j]);
      o[j] = wmma_f32_16x16x32(pf1, vb1, o[j]);
    }
  }

  // ---- normalize and store y[n][s][h*64+dh] (f16, heads re-interleaved) ----
  const int rr0 = 8 * (lane >> 4), cc = lane & 15;
#pragma unroll
  for (int r = 0; r < 8; ++r) {
    const float inv = 1.f / l_run[r];
    const int s = sblk * BR + w * 16 + r + rr0;
#pragma unroll
    for (int j = 0; j < 4; ++j) {
      const int col = h * DHd + j * 16 + cc;
      y[((size_t)(n * Sd) + s) * Dd + col] = (_Float16)(o[j][r] * inv);
    }
  }
}

// ---------------------------------------------------------------------------
// Launch: 3 projection GEMMs -> flash attention -> output projection.
// Workspace (f16): q | k | v | y, 4 x 8 MB = 32 MB.
// ---------------------------------------------------------------------------
extern "C" void kernel_launch(void* const* d_in, const int* in_sizes, int n_in,
                              void* d_out, int out_size, void* d_ws, size_t ws_size,
                              hipStream_t stream) {
  const float* query = (const float*)d_in[0];
  const float* key   = (const float*)d_in[1];
  const float* value = (const float*)d_in[2];
  const float* Wq = (const float*)d_in[3];
  const float* bq = (const float*)d_in[4];
  const float* Wk = (const float*)d_in[5];
  const float* bk = (const float*)d_in[6];
  const float* Wv = (const float*)d_in[7];
  const float* bv = (const float*)d_in[8];
  const float* Wo = (const float*)d_in[9];
  const float* bo = (const float*)d_in[10];
  float* out = (float*)d_out;

  const int Nb = 4, S = 2048, D = 512;
  const int M = Nb * S;                     // 8192 rows
  const size_t tokElems = (size_t)M * D;    // 4,194,304

  _Float16* qws = (_Float16*)d_ws;
  _Float16* kws = qws + tokElems;
  _Float16* vws = kws + tokElems;
  _Float16* yws = vws + tokElems;

  dim3 gg(D / 64, M / 128);  // 8 x 64 blocks
  gemm_bias_wmma<float, _Float16><<<gg, 256, 0, stream>>>(query, Wq, bq, qws, M, D, D);
  gemm_bias_wmma<float, _Float16><<<gg, 256, 0, stream>>>(key,   Wk, bk, kws, M, D, D);
  gemm_bias_wmma<float, _Float16><<<gg, 256, 0, stream>>>(value, Wv, bv, vws, M, D, D);

  flash_attn_wmma<<<dim3(Nb * 8 * (S / 128)), 256, 0, stream>>>(qws, kws, vws, yws);

  gemm_bias_wmma<_Float16, float><<<gg, 256, 0, stream>>>(yws, Wo, bo, out, M, D, D);
}
